// Attention_88244398063715
// MI455X (gfx1250) — matrix operand order repaired and checked by reading, not compiled, not run on hardware
//
#include <hip/hip_runtime.h>
#include <hip/hip_bf16.h>

#define BB  8
#define NN  2048
#define MM  2048
#define DIN 256
#define DHD 32

typedef __attribute__((ext_vector_type(16))) _Float16 v16h;
typedef __attribute__((ext_vector_type(8)))  _Float16 v8h;
typedef __attribute__((ext_vector_type(8)))  float    v8f;

// SWAPX16: group-of-32 swizzle, xor_mask=0x10, and_mask=0x1f -> offset 0x401F
__device__ __forceinline__ float xor16(float x) {
    return __int_as_float(__builtin_amdgcn_ds_swizzle(__float_as_int(x), 0x401F));
}

// ---------------------------------------------------------------------------
// Kernel 0: convert W (256x32 f32, row-major) -> WT (32x256 f16, [dh][k])
// so proj_wmma can load WMMA B-operands as contiguous 32B chunks.
// ---------------------------------------------------------------------------
__global__ __launch_bounds__(256) void prep_wt(
    const float* __restrict__ Wq, const float* __restrict__ Wk,
    const float* __restrict__ Wv,
    _Float16* __restrict__ wtq, _Float16* __restrict__ wtk,
    _Float16* __restrict__ wtv)
{
    const float* W  = (blockIdx.x == 0) ? Wq  : (blockIdx.x == 1) ? Wk  : Wv;
    _Float16*    WT = (blockIdx.x == 0) ? wtq : (blockIdx.x == 1) ? wtk : wtv;
    #pragma unroll
    for (int i = 0; i < 32; ++i) {
        const int e  = threadIdx.x + 256 * i;   // coalesced read of W
        const int kk = e >> 5, dh = e & 31;
        WT[dh * DIN + kk] = (_Float16)W[e];
    }
}

// ---------------------------------------------------------------------------
// Kernel 1: WMMA projections. One wave per 16-row tile (3072 tiles).
//   Y(16x32) = X(16x256) @ W(256x32); out = tanh(Y + bias)
//   8 K-chunks x 2 dh-halves = 16 x v_wmma_f32_16x16x32_f16 per tile.
//   qh/kh stored row-major f16 [row][dh]; vh stored transposed [b][dh][m].
// ---------------------------------------------------------------------------
__global__ __launch_bounds__(256) void proj_wmma(
    const float* __restrict__ q, const float* __restrict__ k,
    const float* __restrict__ v,
    const float* __restrict__ bq, const float* __restrict__ bk,
    const float* __restrict__ bv,
    const _Float16* __restrict__ wtq, const _Float16* __restrict__ wtk,
    const _Float16* __restrict__ wtv,
    _Float16* __restrict__ qh, _Float16* __restrict__ kh,
    _Float16* __restrict__ vT)
{
    const int lane  = threadIdx.x & 31;
    const int wv    = threadIdx.x >> 5;
    const int lhalf = lane & 15;
    const int hid   = lane >> 4;

    const int tile  = blockIdx.x * 8 + wv;    // 0..3071
    const int mat   = tile >> 10;             // 1024 tiles per matrix
    const int rowg0 = (tile & 1023) * 16;     // global row base = b*2048 + r0

    const float*    x    = (mat == 0) ? q   : (mat == 1) ? k   : v;
    const float*    bias = (mat == 0) ? bq  : (mat == 1) ? bk  : bv;
    const _Float16* WT   = (mat == 0) ? wtq : (mat == 1) ? wtk : wtv;

    // accumulators initialized with bias (C/D layout: dh = 16h + lhalf, all VGPRs)
    const float b0 = bias[lhalf];
    const float b1 = bias[16 + lhalf];
    v8f acc0, acc1;
    #pragma unroll
    for (int j = 0; j < 8; ++j) { acc0[j] = b0; acc1[j] = b1; }

    // A-operand source row: lane covers k-chunks {0..15} (hid=0) /
    // {16..31} (hid=1) of each 32-wide K chunk
    const float* xr = x + (size_t)(rowg0 + lhalf) * DIN + hid * 16;

    #pragma unroll
    for (int c = 0; c < 8; ++c) {
        const float4* xp = (const float4*)(xr + 32 * c);
        const float4 f0 = xp[0], f1 = xp[1];   // K 32c + hid*16 + {0..7}
        const float4 f2 = xp[2], f3 = xp[3];   // K 32c + hid*16 + 8 + {0..7}
        v16h a;
        a[0]  = (_Float16)f0.x; a[1]  = (_Float16)f0.y;
        a[2]  = (_Float16)f0.z; a[3]  = (_Float16)f0.w;
        a[4]  = (_Float16)f1.x; a[5]  = (_Float16)f1.y;
        a[6]  = (_Float16)f1.z; a[7]  = (_Float16)f1.w;
        a[8]  = (_Float16)f2.x; a[9]  = (_Float16)f2.y;
        a[10] = (_Float16)f2.z; a[11] = (_Float16)f2.w;
        a[12] = (_Float16)f3.x; a[13] = (_Float16)f3.y;
        a[14] = (_Float16)f3.z; a[15] = (_Float16)f3.w;

        // B-operands from WT[dh][k]: lane n -> column dh, contiguous 16 halves
        const v16h bm0 = *(const v16h*)(WT + (size_t)lhalf * DIN + 32 * c + hid * 16);
        const v16h bm1 = *(const v16h*)(WT + (size_t)(16 + lhalf) * DIN + 32 * c + hid * 16);

        acc0 = __builtin_amdgcn_wmma_f32_16x16x32_f16(
            false, a, false, bm0, (short)0, acc0, false, false);
        acc1 = __builtin_amdgcn_wmma_f32_16x16x32_f16(
            false, a, false, bm1, (short)0, acc1, false, false);
    }

    // epilogue: tanh + store (C/D layout: VGPR j -> row j + 8*hid)
    if (mat < 2) {
        _Float16* dst = (mat == 0) ? qh : kh;
        #pragma unroll
        for (int j = 0; j < 8; ++j) {
            const size_t row = (size_t)(rowg0 + j + 8 * hid);
            dst[row * DHD + lhalf]      = (_Float16)tanhf(acc0[j]);
            dst[row * DHD + 16 + lhalf] = (_Float16)tanhf(acc1[j]);
        }
    } else {
        const int b = rowg0 >> 11;
        #pragma unroll
        for (int j = 0; j < 8; ++j) {
            const int r = (rowg0 + j + 8 * hid) & 2047;
            vT[((size_t)b * DHD + lhalf) * MM + r]      = (_Float16)tanhf(acc0[j]);
            vT[((size_t)b * DHD + 16 + lhalf) * MM + r] = (_Float16)tanhf(acc1[j]);
        }
    }
}

// ---------------------------------------------------------------------------
// Kernel 2: flash attention, one wave per 16-row Q tile.
// S^T = K_tile(16x32dh) x Q^T(32dh x 16rows)  via v_wmma_f32_16x16x32_f16
// online softmax per q-row (lane pair, SWAPX16 combine)
// O^T += V^T(16dh x 32keys) x P^T(32keys x 16rows)
// ---------------------------------------------------------------------------
__global__ __launch_bounds__(256) void fattn_kernel(
    const _Float16* __restrict__ qh, const _Float16* __restrict__ kh,
    const _Float16* __restrict__ vT, float* __restrict__ out)
{
    const int lane  = threadIdx.x & 31;
    const int wv    = threadIdx.x >> 5;
    const int lhalf = lane & 15;
    const int hid   = lane >> 4;                 // 0 or 1 (half-wave)

    const int tile  = blockIdx.x * 8 + wv;       // 0..1023
    const int b     = tile >> 7;                 // 128 q-tiles per batch
    const int qrow0 = (tile & 127) * 16;

    // Q^T B-operand: lane n(<16) -> qh[row n][dh 0..15]; lane n+16 -> dh 16..31
    const _Float16* qbase =
        qh + ((size_t)(b * NN + qrow0 + lhalf)) * DHD + hid * 16;
    const v16h bq16 = *(const v16h*)qbase;

    const _Float16* kbB = kh + (size_t)b * MM * DHD;   // [m][dh]
    const _Float16* vbB = vT + (size_t)b * DHD * MM;   // [dh][m]

    v8f o0 = {}, o1 = {};
    float m = -1.0e30f, l = 0.0f;

    for (int kt = 0; kt < MM; kt += 32) {
        // ---- K A-operands (A layout halves {0..15}/{16..31})
        v16h a0, a1;
        {
            const _Float16* p0 = kbB + (size_t)(kt + lhalf) * DHD + hid * 16;
            v8h lo  = *(const v8h*)p0;
            v8h hi  = *(const v8h*)(p0 + 8);
            const _Float16* p1 = p0 + 16 * DHD;
            v8h lo1 = *(const v8h*)p1;
            v8h hi1 = *(const v8h*)(p1 + 8);
            #pragma unroll
            for (int i = 0; i < 8; ++i) {
                a0[i] = lo[i];  a0[i + 8] = hi[i];
                a1[i] = lo1[i]; a1[i + 8] = hi1[i];
            }
        }
        // prefetch next K tile (clamped; locality 3 -> near caches)
        {
            const int ktn = (kt + 32 < MM) ? (kt + 32) : kt;
            __builtin_prefetch(kbB + (size_t)(ktn + lhalf) * DHD, 0, 3);
        }

        v8f zero = {};
        v8f s0 = __builtin_amdgcn_wmma_f32_16x16x32_f16(
            false, a0, false, bq16, (short)0, zero, false, false);
        v8f s1 = __builtin_amdgcn_wmma_f32_16x16x32_f16(
            false, a1, false, bq16, (short)0, zero, false, false);
        // s0: lane<16 keys kt+{0..7}, lane>=16 keys kt+{8..15}; s1: +16

        // ---- online softmax, per q-row (lane pair n, n+16)
        float tmax = s0[0];
        #pragma unroll
        for (int j = 0; j < 8; ++j) {
            tmax = fmaxf(tmax, s0[j]);
            tmax = fmaxf(tmax, s1[j]);
        }
        tmax = fmaxf(tmax, xor16(tmax));
        const float mnew = fmaxf(m, tmax);
        const float corr = __expf(m - mnew);
        m = mnew;

        float p0v[8], p1v[8], psum = 0.0f;
        #pragma unroll
        for (int j = 0; j < 8; ++j) {
            p0v[j] = __expf(s0[j] - mnew);
            p1v[j] = __expf(s1[j] - mnew);
            psum += p0v[j] + p1v[j];
        }
        psum += xor16(psum);
        l = l * corr + psum;

        #pragma unroll
        for (int j = 0; j < 8; ++j) { o0[j] *= corr; o1[j] *= corr; }

        // ---- build P^T f16 B-operand (rel keys: lane<16 -> 0..15, lane>=16 -> 16..31)
        v16h pb;
        #pragma unroll
        for (int j = 0; j < 8; ++j) {
            const float x0 = xor16(p0v[j]);   // partner's s0 exp values
            const float x1 = xor16(p1v[j]);   // partner's s1 exp values
            pb[j]     = (_Float16)((hid == 0) ? p0v[j] : x1);
            pb[j + 8] = (_Float16)((hid == 0) ? x0 : p1v[j]);
        }

        // ---- V^T A-operands: dh rows 0..15 (av0), 16..31 (av1), keys kt..kt+31
        v16h av0, av1;
        {
            const _Float16* pv0 = vbB + (size_t)lhalf * MM + kt + hid * 16;
            v8h lo  = *(const v8h*)pv0;
            v8h hi  = *(const v8h*)(pv0 + 8);
            const _Float16* pv1 = pv0 + 16 * MM;
            v8h lo1 = *(const v8h*)pv1;
            v8h hi1 = *(const v8h*)(pv1 + 8);
            #pragma unroll
            for (int i = 0; i < 8; ++i) {
                av0[i] = lo[i];  av0[i + 8] = hi[i];
                av1[i] = lo1[i]; av1[i + 8] = hi1[i];
            }
        }

        o0 = __builtin_amdgcn_wmma_f32_16x16x32_f16(
            false, av0, false, pb, (short)0, o0, false, false);
        o1 = __builtin_amdgcn_wmma_f32_16x16x32_f16(
            false, av1, false, pb, (short)0, o1, false, false);
    }

    // ---- normalize and scatter O (out[b][n][dh], f32)
    const float invl = 1.0f / l;
    float* orow = out + ((size_t)(b * NN + qrow0 + lhalf)) * DHD;
    const int dh0 = hid * 8;
    #pragma unroll
    for (int j = 0; j < 8; ++j) {
        orow[dh0 + j]      = o0[j] * invl;   // dh 0..15 half
        orow[16 + dh0 + j] = o1[j] * invl;   // dh 16..31 half
    }
}

// ---------------------------------------------------------------------------
extern "C" void kernel_launch(void* const* d_in, const int* in_sizes, int n_in,
                              void* d_out, int out_size, void* d_ws, size_t ws_size,
                              hipStream_t stream)
{
    const float* q  = (const float*)d_in[0];
    const float* k  = (const float*)d_in[1];
    const float* v  = (const float*)d_in[2];
    const float* Wq = (const float*)d_in[3];
    const float* bq = (const float*)d_in[4];
    const float* Wk = (const float*)d_in[5];
    const float* bk = (const float*)d_in[6];
    const float* Wv = (const float*)d_in[7];
    const float* bv = (const float*)d_in[8];
    float* out = (float*)d_out;

    _Float16* qh  = (_Float16*)d_ws;                       // 1 MB
    _Float16* kh  = qh  + (size_t)BB * NN * DHD;           // 1 MB
    _Float16* vT  = kh  + (size_t)BB * MM * DHD;           // 1 MB
    _Float16* wtq = vT  + (size_t)BB * DHD * MM;           // 16 KB
    _Float16* wtk = wtq + (size_t)DIN * DHD;               // 16 KB
    _Float16* wtv = wtk + (size_t)DIN * DHD;               // 16 KB

    prep_wt<<<3, 256, 0, stream>>>(Wq, Wk, Wv, wtq, wtk, wtv);
    // 3072 projection tiles / 8 waves per block = 384 blocks
    proj_wmma<<<384, 256, 0, stream>>>(q, k, v, bq, bk, bv,
                                       wtq, wtk, wtv, qh, kh, vT);
    // 1024 q-tiles / 8 waves per block = 128 blocks
    fattn_kernel<<<128, 256, 0, stream>>>(qh, kh, vT, out);
}
